// Kmeans_45200235823573
// MI455X (gfx1250) — compile-verified, matrix operand order checked by reading, not run on hardware
//
#include <hip/hip_runtime.h>
#include <hip/hip_bf16.h>

#define CB_DIM  1280
#define CB_SIZE 8192
#define NROWS   16384      // 8 * 2048
#define MTILE   64
#define NBLK    128        // columns swept per block iteration: 4 N-waves * 32

typedef __attribute__((ext_vector_type(16))) _Float16 v16h;
typedef __attribute__((ext_vector_type(8)))  _Float16 v8h;
typedef __attribute__((ext_vector_type(8)))  float    v8f;
typedef __attribute__((ext_vector_type(4)))  unsigned int u32x4;
typedef __attribute__((ext_vector_type(8)))  int i32x8;
typedef __attribute__((ext_vector_type(4)))  int i32x4;
union V16H { v16h v; v8h h[2]; };

// ---------- Kernel 1: L2-normalize each row of x, emit f16 ----------
__global__ void k_normalize(const float* __restrict__ x, _Float16* __restrict__ xn)
{
    __shared__ float red[256];
    const int row = blockIdx.x;
    const int t   = threadIdx.x;
    const float* xr = x + (size_t)row * CB_DIM;
    float v[5];
    float s = 0.f;
#pragma unroll
    for (int j = 0; j < 5; ++j) { v[j] = xr[t + j*256]; s += v[j]*v[j]; }
    red[t] = s; __syncthreads();
    for (int off = 128; off > 0; off >>= 1) {
        if (t < off) red[t] += red[t+off];
        __syncthreads();
    }
    const float scale = 1.f / fmaxf(sqrtf(red[0]), 1e-12f);
    _Float16* xo = xn + (size_t)row * CB_DIM;
#pragma unroll
    for (int j = 0; j < 5; ++j) xo[t + j*256] = (_Float16)(v[j]*scale);
}

// ---------- Kernel 2: transpose codebook (1280 x 8192 f32) -> cbT (8192 x 1280 f16) ----------
__global__ void k_transpose_cb(const float* __restrict__ cb, _Float16* __restrict__ cbT)
{
    __shared__ float tile[32][33];
    const int n0 = blockIdx.x * 32;
    const int k0 = blockIdx.y * 32;
    const int tx = threadIdx.x & 31;
    const int ty = threadIdx.x >> 5;         // 0..7
#pragma unroll
    for (int r = 0; r < 4; ++r)
        tile[ty + r*8][tx] = cb[(size_t)(k0 + ty + r*8) * CB_SIZE + n0 + tx];
    __syncthreads();
#pragma unroll
    for (int r = 0; r < 4; ++r)
        cbT[(size_t)(n0 + ty + r*8) * CB_DIM + k0 + tx] = (_Float16)tile[tx][ty + r*8];
}

// ---------- Kernel 3: per-code squared norms ----------
__global__ void k_colnorm(const float* __restrict__ cb, float* __restrict__ cnorm)
{
    const int n = blockIdx.x * 256 + threadIdx.x;
    float s = 0.f;
    for (int k = 0; k < CB_DIM; ++k) {
        float v = cb[(size_t)k * CB_SIZE + n];
        s += v * v;
    }
    cnorm[n] = s;
}

// ---------- helper: fused (min, argmin) update over the 8 row-slots of one C tile ----------
__device__ __forceinline__ void upd(float* best, int* bidx, float cn, const v8f& a, int n)
{
#pragma unroll
    for (int r = 0; r < 8; ++r) {
        float d = cn - 2.f * a[r];         // ||x_hat||^2 == 1 is a constant shift
        if (d < best[r]) { best[r] = d; bidx[r] = n; }
    }
}

// ---------- Kernel 4: WMMA GEMM + fused argmin ----------
// Block tile M=64. 8 waves = 2 (M-pair: rows wmp*32..+31) x 4 (N: wn*32 within NBLK).
// Each wave computes a 32x32 tile: 2 A frags x 2 B frags -> 4 wmma per K-step,
// i.e. 1 LDS b128 + 1 global b128 per wmma.
// A tile (64x1280 f16 = 160 KB) is DMA'd into LDS once via the Tensor Data Mover.
__global__ void k_argmin_wmma(const _Float16* __restrict__ xn,
                              const _Float16* __restrict__ cbT,
                              const float*    __restrict__ cnorm,
                              int*            __restrict__ inds,
                              float*          __restrict__ out_inds)
{
    extern __shared__ _Float16 ldsA[];        // MTILE * CB_DIM halves = 160 KB
    __shared__ float sMin[MTILE][4];
    __shared__ int   sIdx[MTILE][4];

    const int tid  = threadIdx.x;
    const int lane = tid & 31;
    const int wid  = tid >> 5;                // 0..7
    const int wmp  = wid >> 2;                // 0..1 -> rows wmp*32 .. +31
    const int wn   = wid & 3;                 // 0..3 -> cols wn*32 within NBLK
    const int row0 = blockIdx.x * MTILE;

    // ---- TDM stage: one tensor_load_to_lds per block (wave 0 only) ----
    if (tid < 32) {
        const unsigned ldsOff = (unsigned)(size_t)ldsA;   // LDS byte offset
        const unsigned long long ga =
            (unsigned long long)(size_t)(xn + (size_t)row0 * CB_DIM);
        u32x4 g0; i32x8 g1;
        i32x4 g2 = {0,0,0,0};
        i32x4 g3 = {0,0,0,0};
        i32x8 g4 = {0,0,0,0,0,0,0,0};
        // D# group 0: count=1 | lds_addr | global_addr[56:0] | type=2
        g0[0] = 1u;
        g0[1] = ldsOff;
        g0[2] = (unsigned)ga;
        g0[3] = (unsigned)(ga >> 32) | (2u << 30);
        // D# group 1: data_size=2B, tensor 1280 x 64, tile 1280 x 64, stride0=1280
        g1[0] = 1 << 16;                       // data_size = 1 -> 2 bytes
        g1[1] = (int)((unsigned)CB_DIM << 16); // tensor_dim0[15:0] = 1280
        g1[2] = (int)((unsigned)MTILE << 16);  // tensor_dim0[31:16]=0 | tensor_dim1[15:0]=64
        g1[3] = (int)((unsigned)CB_DIM << 16); // tensor_dim1[31:16]=0 | tile_dim0=1280
        g1[4] = MTILE;                         // tile_dim1=64, tile_dim2=0
        g1[5] = CB_DIM;                        // tensor_dim0_stride = 1280
        g1[6] = 0;
        g1[7] = 0;
        __builtin_amdgcn_tensor_load_to_lds(g0, g1, g2, g3, g4, 0);
        __builtin_amdgcn_s_wait_tensorcnt(0);
    }
    __syncthreads();

    // 16-bit fragment layout (ISA 7.12.2): lanes 0-15 carry K {0..7,16..23},
    // lanes 16-31 carry K {8..15,24..31}; lane%16 = M row (A) / N col (B).
    const int koff = (lane < 16) ? 0 : 8;
    const int slot = lane & 15;
    const _Float16* arow0 = ldsA + (size_t)(wmp*32 + slot) * CB_DIM;
    const _Float16* arow1 = arow0 + (size_t)16 * CB_DIM;

    float best[2][8];
    int   bidx[2][8];
#pragma unroll
    for (int g = 0; g < 2; ++g)
#pragma unroll
        for (int r = 0; r < 8; ++r) { best[g][r] = 3.4e38f; bidx[g][r] = 0; }

    for (int n0 = 0; n0 < CB_SIZE; n0 += NBLK) {
        const int nA = n0 + wn*32 + slot;      // column owned by this lane, subtile 0
        const int nB = nA + 16;                // subtile 1
        const _Float16* brow0 = cbT + (size_t)nA * CB_DIM;
        const _Float16* brow1 = cbT + (size_t)nB * CB_DIM;
        v8f acc00 = {}, acc01 = {}, acc10 = {}, acc11 = {};
#pragma unroll 2
        for (int k0 = 0; k0 < CB_DIM; k0 += 32) {
            V16H a0, a1, b0, b1;
            a0.h[0] = *(const v8h*)(arow0 + k0 + koff);
            a0.h[1] = *(const v8h*)(arow0 + k0 + koff + 16);
            a1.h[0] = *(const v8h*)(arow1 + k0 + koff);
            a1.h[1] = *(const v8h*)(arow1 + k0 + koff + 16);
            b0.h[0] = *(const v8h*)(brow0 + k0 + koff);
            b0.h[1] = *(const v8h*)(brow0 + k0 + koff + 16);
            b1.h[0] = *(const v8h*)(brow1 + k0 + koff);
            b1.h[1] = *(const v8h*)(brow1 + k0 + koff + 16);
            acc00 = __builtin_amdgcn_wmma_f32_16x16x32_f16(
                        false, a0.v, false, b0.v, (short)0, acc00, false, false);
            acc01 = __builtin_amdgcn_wmma_f32_16x16x32_f16(
                        false, a0.v, false, b1.v, (short)0, acc01, false, false);
            acc10 = __builtin_amdgcn_wmma_f32_16x16x32_f16(
                        false, a1.v, false, b0.v, (short)0, acc10, false, false);
            acc11 = __builtin_amdgcn_wmma_f32_16x16x32_f16(
                        false, a1.v, false, b1.v, (short)0, acc11, false, false);
        }
        const float cnA = cnorm[nA];
        const float cnB = cnorm[nB];
        upd(best[0], bidx[0], cnA, acc00, nA);   // ascending n keeps first-min tie-break
        upd(best[0], bidx[0], cnB, acc01, nB);
        upd(best[1], bidx[1], cnA, acc10, nA);
        upd(best[1], bidx[1], cnB, acc11, nB);
    }

    // C/D layout: lane%16 = column, lane/16 picks rows {r} vs {8+r}.
    // Reduce min+argmin across the 16 column-lanes of each half-wave.
#pragma unroll
    for (int g = 0; g < 2; ++g)
#pragma unroll
    for (int r = 0; r < 8; ++r) {
#pragma unroll
        for (int off = 8; off > 0; off >>= 1) {
            float ob = __shfl_xor(best[g][r], off, 16);
            int   oi = __shfl_xor(bidx[g][r], off, 16);
            if (ob < best[g][r] || (ob == best[g][r] && oi < bidx[g][r])) {
                best[g][r] = ob; bidx[g][r] = oi;
            }
        }
    }
    if ((lane & 15) == 0) {
        const int hhi = (lane >= 16) ? 8 : 0;
#pragma unroll
        for (int g = 0; g < 2; ++g) {
            const int rbase = wmp*32 + g*16 + hhi;
#pragma unroll
            for (int r = 0; r < 8; ++r) {
                sMin[rbase + r][wn] = best[g][r];
                sIdx[rbase + r][wn] = bidx[g][r];
            }
        }
    }
    __syncthreads();
    if (tid < MTILE) {
        float bv = sMin[tid][0]; int bi = sIdx[tid][0];
#pragma unroll
        for (int c = 1; c < 4; ++c) {
            float v = sMin[tid][c]; int i = sIdx[tid][c];
            if (v < bv || (v == bv && i < bi)) { bv = v; bi = i; }
        }
        inds[row0 + tid]     = bi;
        out_inds[row0 + tid] = (float)bi;    // tuple output 0, flattened as float
    }
}

// ---------- Kernel 5: gather quantized = codebook[:, inds]  (shape 1280 x 16384) ----------
__global__ void k_gather(const float* __restrict__ cb, const int* __restrict__ inds,
                         float* __restrict__ outq)
{
    const int i = blockIdx.x * 256 + threadIdx.x;   // 0..16383
    const int k = blockIdx.y;                       // 0..1279
    outq[(size_t)k * NROWS + i] = cb[(size_t)k * CB_SIZE + inds[i]];
}

extern "C" void kernel_launch(void* const* d_in, const int* in_sizes, int n_in,
                              void* d_out, int out_size, void* d_ws, size_t ws_size,
                              hipStream_t stream)
{
    const float* x  = (const float*)d_in[0];   // (8,2048,1280) f32
    const float* cb = (const float*)d_in[1];   // (1280,8192)  f32

    // workspace layout (bytes)
    char* ws = (char*)d_ws;
    _Float16* xn   = (_Float16*)(ws);                                // 16384*1280*2 = 41,943,040
    _Float16* cbT  = (_Float16*)(ws + 41943040ULL);                  //  8192*1280*2 = 20,971,520
    float*    cnm  = (float*)   (ws + 62914560ULL);                  //  8192*4
    int*      inds = (int*)     (ws + 62947328ULL);                  // 16384*4

    float* out_inds = (float*)d_out;            // first 16384 floats: indices
    float* outq     = (float*)d_out + NROWS;    // then 1280*16384 floats: quantized

    k_normalize   <<<NROWS, 256, 0, stream>>>(x, xn);
    k_transpose_cb<<<dim3(CB_SIZE/32, CB_DIM/32), 256, 0, stream>>>(cb, cbT);
    k_colnorm     <<<CB_SIZE/256, 256, 0, stream>>>(cb, cnm);
    k_argmin_wmma <<<NROWS/MTILE, 256, (size_t)MTILE*CB_DIM*sizeof(_Float16), stream>>>(
                      xn, cbT, cnm, inds, out_inds);
    k_gather      <<<dim3(NROWS/256, CB_DIM), 256, 0, stream>>>(cb, inds, outq);
}
